// CustomLSTM_24438363914856
// MI455X (gfx1250) — matrix operand order repaired and checked by reading, not compiled, not run on hardware
//
#include <hip/hip_runtime.h>
#include <hip/hip_bf16.h>

// ---------------------------------------------------------------------------
// CDNA5 (gfx1250) LSTM:
//   - one-time f32->bf16 conversion / transposition of x, kernel, rkernel
//   - bf16 WMMA input-projection GEMM (double-buffered async-to-LDS tiles)
//   - 512 sequential bf16 WMMA recurrent steps (L2-resident bf16 weights,
//     async-to-LDS staging, fused clip/sigmoid/tanh gate epilogue)
// ---------------------------------------------------------------------------

typedef __bf16 bf16;
typedef bf16  v16bf __attribute__((ext_vector_type(16)));
typedef bf16  v8bf  __attribute__((ext_vector_type(8)));
typedef bf16  v4bf  __attribute__((ext_vector_type(4)));
typedef float v8f   __attribute__((ext_vector_type(8)));

#define LSTM_B   64
#define LSTM_T   512
#define LSTM_I   1024
#define LSTM_H   1024
#define LSTM_G   4096   // 4*H
#define LSTM_M   32768  // B*T
#define CLIP_V   3.0f

#define LDT 40          // LDS row stride in bf16 elems (32 + 8 pad, keeps 16B align)

// ---- CDNA5 async copy global -> LDS (ASYNCcnt-tracked) --------------------
__device__ __forceinline__ void async_copy_b128(void* lds, const void* gptr) {
    unsigned l = (unsigned)(size_t)lds;                 // low 32 bits = LDS offset
    unsigned long long g = (unsigned long long)(size_t)gptr;
    asm volatile("global_load_async_to_lds_b128 %0, %1, off"
                 :: "v"(l), "v"(g)
                 : "memory");
}

#if __has_builtin(__builtin_amdgcn_s_wait_asynccnt)
#define WAIT_ASYNC(n) __builtin_amdgcn_s_wait_asynccnt(n)
#else
#define WAIT_ASYNC(n) asm volatile("s_wait_asynccnt %0" :: "n"(n) : "memory")
#endif

// ---- fragment loaders -----------------------------------------------------
// A (16x32 bf16, MxK): lane L holds row m=L&15; hi=L>>4 selects K-halves:
//   halves 0..7  = K[hi*8 .. hi*8+7],  halves 8..15 = K[hi*8+16 .. hi*8+23]
__device__ __forceinline__ v16bf load_frag_a(const bf16* base, int lane) {
    const int m  = lane & 15;
    const int hi = lane >> 4;
    const bf16* p = base + m * LDT + hi * 8;
    v8bf lo = *(const v8bf*)(p);
    v8bf hv = *(const v8bf*)(p + 16);
    v16bf r;
#pragma unroll
    for (int i = 0; i < 8; ++i) { r[i] = lo[i]; r[i + 8] = hv[i]; }
    return r;
}

// B (32x16 bf16, KxN) from K-contiguous (transposed) LDS tile [n][k]:
// lane L holds col n=L&15; hi=L>>4: halves 0..7 = K[hi*8 .. hi*8+7], halves 8..15 = K[hi*8+16 .. hi*8+23]
__device__ __forceinline__ v16bf load_frag_b(const bf16* base, int lane) {
    const int n  = lane & 15;
    const int hi = lane >> 4;
    const bf16* p = base + n * LDT + hi * 8;
    v8bf lo = *(const v8bf*)(p);
    v8bf hv = *(const v8bf*)(p + 16);
    v16bf r;
#pragma unroll
    for (int i = 0; i < 8; ++i) { r[i] = lo[i]; r[i + 8] = hv[i]; }
    return r;
}

__device__ __forceinline__ float clipf3(float v) {
    return fminf(CLIP_V, fmaxf(-CLIP_V, v));
}
__device__ __forceinline__ float sigmoidf(float x) {
    return 1.0f / (1.0f + __expf(-x));
}

// ---------------------------------------------------------------------------
// Pre-pass A: straight f32 -> bf16 (x)
// ---------------------------------------------------------------------------
__global__ __launch_bounds__(256) void f32_to_bf16(
    const float* __restrict__ src, bf16* __restrict__ dst, size_t nquads)
{
    const size_t i = (size_t)blockIdx.x * 256 + threadIdx.x;
    if (i < nquads) {
        const float4 f = ((const float4*)src)[i];
        v4bf o;
        o[0] = (bf16)f.x; o[1] = (bf16)f.y; o[2] = (bf16)f.z; o[3] = (bf16)f.w;
        ((v4bf*)dst)[i] = o;
    }
}

// ---------------------------------------------------------------------------
// Pre-pass B: transpose [K=1024][N=4096] f32 -> [N=4096][K=1024] bf16
// (used for both `kernel` and `recurrent_kernel`; makes B-matrix K-contiguous)
// ---------------------------------------------------------------------------
__global__ __launch_bounds__(256) void transpose_f32_to_bf16(
    const float* __restrict__ src, bf16* __restrict__ dst)
{
    __shared__ bf16 tile[32][33];
    const int k0 = blockIdx.x * 32;     // K dim (1024/32)
    const int n0 = blockIdx.y * 32;     // N dim (4096/32)
    const int r  = threadIdx.x >> 3;    // 0..31
    const int c4 = (threadIdx.x & 7) << 2;
    const float4 f = *(const float4*)(src + (size_t)(k0 + r) * LSTM_G + n0 + c4);
    tile[r][c4 + 0] = (bf16)f.x;
    tile[r][c4 + 1] = (bf16)f.y;
    tile[r][c4 + 2] = (bf16)f.z;
    tile[r][c4 + 3] = (bf16)f.w;
    __syncthreads();
    const int n  = threadIdx.x >> 3;
    const int k4 = (threadIdx.x & 7) << 2;
    v4bf o;
    o[0] = tile[k4 + 0][n];
    o[1] = tile[k4 + 1][n];
    o[2] = tile[k4 + 2][n];
    o[3] = tile[k4 + 3][n];
    *(v4bf*)(dst + (size_t)(n0 + n) * 1024 + k0 + k4) = o;
}

// ---------------------------------------------------------------------------
// Kernel 1: x_proj[M, N=4H] = Xb[M, K] * Wb^T   (bf16 WMMA, f32 acc)
// Xb: [M][1024] bf16 row-major;  Wb: [4096][1024] bf16 (K-contiguous per col)
// block = 256 threads (8 waves), tile 128x128, K-chunk 32, double-buffered
// async-to-LDS staging.
// ---------------------------------------------------------------------------
__global__ __launch_bounds__(256) void xproj_gemm(
    const bf16* __restrict__ Xb, const bf16* __restrict__ Wb,
    float* __restrict__ P)
{
    __shared__ bf16 As[2][128][LDT];
    __shared__ bf16 Bs[2][128][LDT];

    const int tid  = threadIdx.x;
    const int lane = tid & 31;
    const int wave = tid >> 5;
    const int row0 = blockIdx.y * 128;
    const int col0 = blockIdx.x * 128;
    const int wr   = (wave >> 1) * 32;
    const int wc   = (wave & 1) * 64;

    v8f acc[2][4] = {};

    // copy geometry: 128 rows x 32 bf16 per tile; 2 threads/row, 32B each
    const int trow = tid >> 1;
    const int tseg = (tid & 1) << 4;    // half-element offset 0 or 16

    const bf16* gA = Xb + (size_t)(row0 + trow) * 1024 + tseg;
    const bf16* gB = Wb + (size_t)(col0 + trow) * 1024 + tseg;

#define GEMM_ISSUE(k0, buf)                                        \
    do {                                                           \
        async_copy_b128(&As[buf][trow][tseg],     gA + (k0));      \
        async_copy_b128(&As[buf][trow][tseg + 8], gA + (k0) + 8);  \
        async_copy_b128(&Bs[buf][trow][tseg],     gB + (k0));      \
        async_copy_b128(&Bs[buf][trow][tseg + 8], gB + (k0) + 8);  \
    } while (0)

    GEMM_ISSUE(0, 0);
    for (int ic = 0; ic < 32; ++ic) {
        if (ic + 1 < 32) {
            GEMM_ISSUE((ic + 1) * 32, (ic + 1) & 1);
            WAIT_ASYNC(4);              // chunk ic's 4 copies complete
        } else {
            WAIT_ASYNC(0);
        }
        __syncthreads();

        const int buf = ic & 1;
        v16bf afr[2], bfr[4];
#pragma unroll
        for (int i = 0; i < 2; ++i) afr[i] = load_frag_a(&As[buf][wr + i * 16][0], lane);
#pragma unroll
        for (int j = 0; j < 4; ++j) bfr[j] = load_frag_b(&Bs[buf][wc + j * 16][0], lane);
#pragma unroll
        for (int i = 0; i < 2; ++i)
#pragma unroll
            for (int j = 0; j < 4; ++j)
                acc[i][j] = __builtin_amdgcn_wmma_f32_16x16x32_bf16(
                    false, afr[i], false, bfr[j], (short)0, acc[i][j], false, false);
        __syncthreads();
    }
#undef GEMM_ISSUE

    // C/D layout: VGPR r -> row = r + (lane>>4)*8, col = lane&15
    const int hi = lane >> 4;
    const int nl = lane & 15;
#pragma unroll
    for (int i = 0; i < 2; ++i)
#pragma unroll
        for (int j = 0; j < 4; ++j) {
            const size_t rbase = (size_t)(row0 + wr + i * 16 + hi * 8);
            const int    cidx  = col0 + wc + j * 16 + nl;
#pragma unroll
            for (int r = 0; r < 8; ++r)
                P[(rbase + r) * LSTM_G + cidx] = acc[i][j][r];
        }
}

// ---------------------------------------------------------------------------
// Kernel 2: one LSTM timestep.
// Rb: [4096][1024] bf16 (pre-transposed recurrent weights, L2-resident 8MB).
// Each block owns 32 h-columns; computes 4 gate tiles (64x32) via WMMA over
// K=1024 with double-buffered async-to-LDS h/R tiles, then fused epilogue.
// ---------------------------------------------------------------------------
__global__ __launch_bounds__(256) void lstm_step(
    const float* __restrict__ xproj, const bf16* __restrict__ Rb,
    const float* __restrict__ bi, const float* __restrict__ bfg,
    const float* __restrict__ bc, const float* __restrict__ bo,
    const bf16* __restrict__ hin, bf16* __restrict__ hout,
    float* __restrict__ cstate, float* __restrict__ outseq, int t)
{
    __shared__ bf16  Hs[2][64][LDT];     // h tile (A): 64 rows x 32 k
    __shared__ bf16  Rs[2][128][LDT];    // R tile (B): row = gate*32+col, [k]
    __shared__ float Gs[4][64][32];      // gate pre-activations

    const int tid  = threadIdx.x;
    const int lane = tid & 31;
    const int wave = tid >> 5;
    const int n0   = blockIdx.x * 32;
    const int g    = wave >> 1;
    const int sc   = (wave & 1) * 16;

    v8f acc[4] = {};

    // h copy geometry: 64 rows x 32 bf16; 4 threads/row, 16B each
    const int hr = tid >> 2;
    const int hc = (tid & 3) << 3;
    const bf16* gH = hin + (size_t)hr * LSTM_H + hc;
    // R copy geometry: 128 rows x 32 bf16; 2 threads/row, 32B each
    const int rrow = tid >> 1;
    const int rseg = (tid & 1) << 4;
    const bf16* gR = Rb + ((size_t)(rrow >> 5) * LSTM_H + n0 + (rrow & 31)) * 1024 + rseg;

#define STEP_ISSUE(k0, buf)                                        \
    do {                                                           \
        async_copy_b128(&Hs[buf][hr][hc],         gH + (k0));      \
        async_copy_b128(&Rs[buf][rrow][rseg],     gR + (k0));      \
        async_copy_b128(&Rs[buf][rrow][rseg + 8], gR + (k0) + 8);  \
    } while (0)

    STEP_ISSUE(0, 0);
    for (int ic = 0; ic < 32; ++ic) {
        if (ic + 1 < 32) {
            STEP_ISSUE((ic + 1) * 32, (ic + 1) & 1);
            WAIT_ASYNC(3);
        } else {
            WAIT_ASYNC(0);
        }
        __syncthreads();

        const int buf = ic & 1;
        const v16bf bfr = load_frag_b(&Rs[buf][g * 32 + sc][0], lane);
#pragma unroll
        for (int i = 0; i < 4; ++i) {
            const v16bf afr = load_frag_a(&Hs[buf][i * 16][0], lane);
            acc[i] = __builtin_amdgcn_wmma_f32_16x16x32_bf16(
                false, afr, false, bfr, (short)0, acc[i], false, false);
        }
        __syncthreads();
    }
#undef STEP_ISSUE

    // dump pre-activations to LDS
    const int hi = lane >> 4;
    const int nl = lane & 15;
#pragma unroll
    for (int i = 0; i < 4; ++i)
#pragma unroll
        for (int r = 0; r < 8; ++r)
            Gs[g][i * 16 + hi * 8 + r][sc + nl] = acc[i][r];
    __syncthreads();

    // fused gate epilogue: 64 rows x 32 cols = 2048 elements, 8 per thread
#pragma unroll
    for (int e = 0; e < 8; ++e) {
        const int idx = e * 256 + tid;
        const int b = idx >> 5;
        const int c = idx & 31;
        const int n = n0 + c;
        const size_t xo = ((size_t)b * LSTM_T + t) * LSTM_G;
        const float pi = Gs[0][b][c] + xproj[xo + n]              + bi[n];
        const float pf = Gs[1][b][c] + xproj[xo + LSTM_H + n]     + bfg[n];
        const float pc = Gs[2][b][c] + xproj[xo + 2 * LSTM_H + n] + bc[n];
        const float po = Gs[3][b][c] + xproj[xo + 3 * LSTM_H + n] + bo[n];
        const float ig = sigmoidf(clipf3(pi));
        const float fg = sigmoidf(clipf3(pf));
        const float cg = tanhf(clipf3(pc));
        const float og = sigmoidf(clipf3(po));
        const float cold = cstate[b * LSTM_H + n];
        const float cnew = fg * cold + ig * cg;
        const float hnew = og * tanhf(cnew);
        cstate[b * LSTM_H + n] = cnew;
        outseq[((size_t)b * LSTM_T + t) * LSTM_H + n] = hnew;
        hout[b * LSTM_H + n] = (bf16)hnew;
    }
}

// ---------------------------------------------------------------------------
__global__ __launch_bounds__(256) void lstm_init(bf16* h0, float* c0)
{
    const int idx = blockIdx.x * 256 + threadIdx.x;
    if (idx < LSTM_B * LSTM_H) {
        h0[idx] = (bf16)0.0f;
        c0[idx] = 0.0f;
    }
}

__global__ __launch_bounds__(256) void lstm_finalize(
    const float* __restrict__ outseq, const float* __restrict__ cstate,
    float* __restrict__ hlast, float* __restrict__ clast)
{
    const int idx = blockIdx.x * 256 + threadIdx.x;
    if (idx < LSTM_B * LSTM_H) {
        const int b = idx >> 10;
        const int n = idx & (LSTM_H - 1);
        hlast[idx] = outseq[((size_t)b * LSTM_T + (LSTM_T - 1)) * LSTM_H + n];
        clast[idx] = cstate[idx];
    }
}

// ---------------------------------------------------------------------------
extern "C" void kernel_launch(void* const* d_in, const int* in_sizes, int n_in,
                              void* d_out, int out_size, void* d_ws, size_t ws_size,
                              hipStream_t stream)
{
    const float* x      = (const float*)d_in[0];   // [B,T,I]
    const float* kern   = (const float*)d_in[1];   // [I,4H]
    const float* rkern  = (const float*)d_in[2];   // [H,4H]
    const float* bi     = (const float*)d_in[3];
    const float* bfg    = (const float*)d_in[4];
    const float* bc     = (const float*)d_in[5];
    const float* bo     = (const float*)d_in[6];

    float* out    = (float*)d_out;
    float* outseq = out;                                       // [B,T,H]
    float* hlast  = out + (size_t)LSTM_B * LSTM_T * LSTM_H;    // [B,H]
    float* clast  = hlast + (size_t)LSTM_B * LSTM_H;           // [B,H]

    // workspace layout
    char* ws = (char*)d_ws;
    float* xproj = (float*)ws;                       // 512 MB
    ws += (size_t)LSTM_M * LSTM_G * sizeof(float);
    bf16* Xb = (bf16*)ws;                            // 64 MB
    ws += (size_t)LSTM_M * LSTM_I * sizeof(bf16);
    bf16* Wb = (bf16*)ws;                            // 8 MB (transposed)
    ws += (size_t)LSTM_G * LSTM_I * sizeof(bf16);
    bf16* Rb = (bf16*)ws;                            // 8 MB (transposed)
    ws += (size_t)LSTM_G * LSTM_H * sizeof(bf16);
    bf16* h0 = (bf16*)ws;  ws += (size_t)LSTM_B * LSTM_H * sizeof(bf16);
    bf16* h1 = (bf16*)ws;  ws += (size_t)LSTM_B * LSTM_H * sizeof(bf16);
    float* cst = (float*)ws;

    lstm_init<<<(LSTM_B * LSTM_H + 255) / 256, 256, 0, stream>>>(h0, cst);

    // one-time precision/layout conversions
    const size_t xq = (size_t)LSTM_M * LSTM_I / 4;
    f32_to_bf16<<<(unsigned)((xq + 255) / 256), 256, 0, stream>>>(x, Xb, xq);
    dim3 tgrid(LSTM_I / 32, LSTM_G / 32);
    transpose_f32_to_bf16<<<tgrid, 256, 0, stream>>>(kern, Wb);
    transpose_f32_to_bf16<<<tgrid, 256, 0, stream>>>(rkern, Rb);

    dim3 ggrid(LSTM_G / 128, LSTM_M / 128);
    xproj_gemm<<<ggrid, 256, 0, stream>>>(Xb, Wb, xproj);

    for (int t = 0; t < LSTM_T; ++t) {
        const bf16* hin = (t & 1) ? h1 : h0;
        bf16* hout      = (t & 1) ? h0 : h1;
        lstm_step<<<LSTM_H / 32, 256, 0, stream>>>(
            xproj, Rb, bi, bfg, bc, bo, hin, hout, cst, outseq, t);
    }

    lstm_finalize<<<(LSTM_B * LSTM_H + 255) / 256, 256, 0, stream>>>(
        outseq, cst, hlast, clast);
}
